// DyChannel_DW_25177098289281
// MI455X (gfx1250) — matrix operand order repaired and checked by reading, not compiled, not run on hardware
//
#include <hip/hip_runtime.h>
#include <hip/hip_bf16.h>
#include <math.h>

// ---------------------------------------------------------------------------
// DyChannel dynamic grouped conv for MI455X (gfx1250, wave32, WMMA)
// b=32, c=128, h=w=64, groups=16 (8 in / 8 out per group), 3x3, pad 1
// K ordering for the implicit GEMM is TAP-MAJOR: k = tap*8 + ig,
// tap = ky*3+kx (0..8, padded to 12), ig = input channel in group (0..7).
// LDS X tile is channel-innermost so each k-octet is one ds_load_b128.
// ---------------------------------------------------------------------------

typedef _Float16 v16h __attribute__((ext_vector_type(16)));
typedef _Float16 v8h  __attribute__((ext_vector_type(8)));
typedef float    v8f  __attribute__((ext_vector_type(8)));

#define NB   32
#define NC   128
#define HW   64
#define NG   16
#define CG   8           // channels per group (in and out)
#define KPAD 96          // 9 taps * 8 ch = 72, padded to 12 taps * 8 = 96
#define NPAD 16          // 8 out-channels padded to 16
#define XROWS 18         // 16 + halo
#define XCOLS 68         // 64 + halo, padded
#define XTILE (XROWS * XCOLS * CG)   // 9792 halves
#define ZOFF  XTILE                  // 8-half zero region after the tile

#if defined(__gfx1250__) && __has_builtin(__builtin_amdgcn_global_load_async_to_lds_b32)
#define HAVE_ASYNC_LDS 1
#else
#define HAVE_ASYNC_LDS 0
#endif

// ---------------------------------------------------------------------------
// Kernel 1: adaptive avgpool to 16x16 + 1x1 conv (128->16) + ReLU
// ---------------------------------------------------------------------------
__global__ __launch_bounds__(256)
void k_route_pool(const float* __restrict__ x, const float* __restrict__ conv_w,
                  const float* __restrict__ conv_b, float* __restrict__ r) {
    const int b = blockIdx.x;
    const int s = threadIdx.x;        // 0..255
    const int sy = s >> 4, sx = s & 15;
    float acc[16];
#pragma unroll
    for (int o = 0; o < 16; ++o) acc[o] = 0.f;

    const float* xb = x + (size_t)b * NC * HW * HW;
    for (int c = 0; c < NC; ++c) {
        const float* xp = xb + (size_t)c * HW * HW + (sy * 4) * HW + sx * 4;
        float p = 0.f;
#pragma unroll
        for (int dy = 0; dy < 4; ++dy)
#pragma unroll
            for (int dx = 0; dx < 4; ++dx) p += xp[dy * HW + dx];
        p *= (1.0f / 16.0f);
#pragma unroll
        for (int o = 0; o < 16; ++o) acc[o] = fmaf(conv_w[o * NC + c], p, acc[o]);
    }
    float* rb = r + (size_t)b * 4096;
#pragma unroll
    for (int o = 0; o < 16; ++o) rb[o * 256 + s] = fmaxf(acc[o] + conv_b[o], 0.f);
}

// ---------------------------------------------------------------------------
// Kernel 2: FC (4096 -> 512) + ReLU + sigmoid -> rw[b][e*128+co]
// ---------------------------------------------------------------------------
__global__ __launch_bounds__(256)
void k_route_fc(const float* __restrict__ r, const float* __restrict__ fc_w,
                const float* __restrict__ fc_b, float* __restrict__ rw) {
    const int idx = blockIdx.x * 256 + threadIdx.x;   // < 32*512
    const int b = idx >> 9, oc = idx & 511;
    const float4* rr = (const float4*)(r + (size_t)b * 4096);
    const float4* ww = (const float4*)(fc_w + (size_t)oc * 4096);
    float acc = fc_b[oc];
    for (int k = 0; k < 1024; ++k) {
        float4 a = rr[k], w = ww[k];
        acc += a.x * w.x + a.y * w.y + a.z * w.z + a.w * w.w;
    }
    float z = fmaxf(acc, 0.f);
    rw[idx] = 1.f / (1.f + __expf(-z));
}

// ---------------------------------------------------------------------------
// Kernel 3: combine experts into per-(b,g) WMMA B-matrix, f16, tap-major K.
// Bm[(b*16+g)][k][n]: k = tap*8+ig (zero for tap>=9), n = out-channel (<8 live)
// ---------------------------------------------------------------------------
__global__ __launch_bounds__(256)
void k_combine(const float* __restrict__ rw, const float* __restrict__ we,
               _Float16* __restrict__ Bm) {
    const int idx = blockIdx.x * 256 + threadIdx.x;  // < 32*16*96*16
    const int n   = idx & 15;
    const int k   = (idx >> 4) % KPAD;
    const int bg  = (idx >> 4) / KPAD;
    const int g   = bg & 15;
    const int b   = bg >> 4;
    float v = 0.f;
    if (n < CG && k < 72) {
        const int tap = k >> 3;       // 0..8
        const int ig  = k & 7;
        const int o   = g * CG + n;
        const float* rwb = rw + (size_t)b * 512;
#pragma unroll
        for (int e = 0; e < 4; ++e)
            v += rwb[e * NC + o] * we[((size_t)(e * NC + o)) * 72 + ig * 9 + tap];
    }
    Bm[idx] = (_Float16)v;
}

// ---------------------------------------------------------------------------
// Kernel 4: implicit-GEMM grouped conv with v_wmma_f32_16x16x32_f16.
// block = (b, g, strip of 16 output rows), 128 threads = 4 waves.
// D tile: M=16 pixels (one row, 16 consecutive x), N=16 (8 live out ch),
// K=96 in three 32-chunks (chunk c = taps 4c..4c+3).
// A fragment per ISA 7.12.2: lane<16 holds K {0..7,16..23}, lane>=16 holds
// K {8..15,24..31} -> each 8-half group = taps (4c+hi) and (4c+2+hi), each
// one aligned b128 load from channel-innermost LDS.
// ---------------------------------------------------------------------------
__global__ __launch_bounds__(128)
void k_dyconv(const float* __restrict__ x, const _Float16* __restrict__ Bm,
              float* __restrict__ out) {
    __shared__ _Float16 ldsX[XTILE + 8];      // [row][col][ig] + zero octet
    __shared__ _Float16 ldsB[KPAD * NPAD];    // 96 x 16, k-major

    const int blk   = blockIdx.x;
    const int strip = blk & 3;
    const int g     = (blk >> 2) & 15;
    const int b     = blk >> 6;
    const int tid   = threadIdx.x;

    // ---- stage B matrix (3 KB) into LDS, async-to-LDS path ----
    const unsigned* bsrc   = (const unsigned*)(Bm + (size_t)(b * NG + g) * (KPAD * NPAD));
    unsigned*       ldsB32 = (unsigned*)ldsB;
#if HAVE_ASYNC_LDS
#pragma unroll
    for (int i = tid; i < (KPAD * NPAD) / 2; i += 128)
        __builtin_amdgcn_global_load_async_to_lds_b32(
            (__attribute__((address_space(1))) int*)(bsrc + i),
            (__attribute__((address_space(3))) int*)(ldsB32 + i), 0, 0);
#else
    for (int i = tid; i < (KPAD * NPAD) / 2; i += 128) ldsB32[i] = bsrc[i];
#endif

    // ---- stage activations (f32 -> f16) with halo; channel-innermost ----
    const float* xg = x + (size_t)(b * NC + g * CG) * (HW * HW);
    for (int i = tid; i < CG * XROWS * XCOLS; i += 128) {
        const int ig  = i / (XROWS * XCOLS);        // coalesced reads along col
        const int rem = i % (XROWS * XCOLS);
        const int row = rem / XCOLS, col = rem % XCOLS;
        const int y  = strip * 16 + row - 1;
        const int xx = col - 1;
        float v = 0.f;
        if ((unsigned)y < (unsigned)HW && (unsigned)xx < (unsigned)HW)
            v = xg[(size_t)ig * (HW * HW) + y * HW + xx];
        ldsX[(row * XCOLS + col) * CG + ig] = (_Float16)v;
    }
    if (tid < 8) ldsX[ZOFF + tid] = (_Float16)0.f;

#if HAVE_ASYNC_LDS
#if __has_builtin(__builtin_amdgcn_s_wait_asynccnt)
    __builtin_amdgcn_s_wait_asynccnt(0);
#else
    asm volatile("s_wait_asynccnt 0" ::: "memory");
#endif
#endif
    __syncthreads();

    const int lane = tid & 31;
    const int wv   = tid >> 5;
    const int m    = lane & 15;           // D row = pixel-in-tile = A row
    const int hi   = (lane >> 4) & 1;     // which K half this lane holds
    const int n    = lane & 15;           // D / B column (output channel)

    union U16 { v16h v; v8h g[2]; _Float16 h[16]; };

    // ---- hoist B fragments into registers (B 32x16 f16 striping:
    // lanes 0-15 hold K {0..7,16..23}, lanes 16-31 hold K {8..15,24..31}, column = lane&15) ----
    U16 fb0, fb1, fb2;
#pragma unroll
    for (int i = 0; i < 16; ++i) {
        const int kb = hi * 8 + (i & 7) + (i >> 3) * 16;
        fb0.h[i] = ldsB[(kb)      * NPAD + n];
        fb1.h[i] = ldsB[(32 + kb) * NPAD + n];
        fb2.h[i] = ldsB[(64 + kb) * NPAD + n];
    }

    // per-lane, loop-invariant base offsets for each tap group
    auto baseOf = [&](int tap) {
        const int ky = tap / 3, kx = tap - 3 * ky;
        return (ky * XCOLS + kx + m) * CG;
    };
    const int bo0 = baseOf(hi);        // chunk0 halves 0..7   (taps 0/1)
    const int bo1 = baseOf(2 + hi);    // chunk0 halves 8..15  (taps 2/3)
    const int bo2 = baseOf(4 + hi);    // chunk1 halves 0..7   (taps 4/5)
    const int bo3 = baseOf(6 + hi);    // chunk1 halves 8..15  (taps 6/7)
    const int bo4 = baseOf(8);         // chunk2 halves 0..7   (tap 8, hi=0 only)
    const v8h z8 = {};

    // 64 pixel tiles per block (16 rows x 4 x-tiles), 16 per wave
#pragma unroll 2
    for (int t = wv; t < 64; t += 4) {
        const int yl = t >> 2;
        const int x0 = (t & 3) << 4;
        const int tb = (yl * XCOLS + x0) * CG;

        U16 a0, a1, a2;
        a0.g[0] = *(const v8h*)(ldsX + tb + bo0);
        a0.g[1] = *(const v8h*)(ldsX + tb + bo1);
        a1.g[0] = *(const v8h*)(ldsX + tb + bo2);
        a1.g[1] = *(const v8h*)(ldsX + tb + bo3);
        a2.g[0] = *(const v8h*)(ldsX + (hi ? ZOFF : tb + bo4));  // tap9 pad -> zeros
        a2.g[1] = z8;                                            // taps 10,11 pad

        v8f c = {};
        c = __builtin_amdgcn_wmma_f32_16x16x32_f16(false, a0.v, false, fb0.v,
                                                   (short)0, c, false, false);
        c = __builtin_amdgcn_wmma_f32_16x16x32_f16(false, a1.v, false, fb1.v,
                                                   (short)0, c, false, false);
        c = __builtin_amdgcn_wmma_f32_16x16x32_f16(false, a2.v, false, fb2.v,
                                                   (short)0, c, false, false);

        // D 16x16 f32: VGPR r -> row m = r + 8*hi, col = lane&15.
        // Rows are consecutive x positions -> two contiguous float4 stores.
        if (n < CG) {
            const int y  = strip * 16 + yl;
            float* po = out + (((size_t)(b * NC + g * CG + n)) * HW + y) * HW + x0 + hi * 8;
            float4 lo  = {c[0], c[1], c[2], c[3]};
            float4 hi4 = {c[4], c[5], c[6], c[7]};
            *(float4*)(po)     = lo;
            *(float4*)(po + 4) = hi4;
        }
    }
}

// ---------------------------------------------------------------------------
// Launcher
// inputs: x, w_experts, conv_w, conv_b, fc_w, fc_b   (all fp32)
// ws: r (32*4096 f32) | rw (32*512 f32) | Bm (32*16*96*16 f16)  ~2.2 MB
// ---------------------------------------------------------------------------
extern "C" void kernel_launch(void* const* d_in, const int* in_sizes, int n_in,
                              void* d_out, int out_size, void* d_ws, size_t ws_size,
                              hipStream_t stream) {
    const float* x         = (const float*)d_in[0];
    const float* w_experts = (const float*)d_in[1];
    const float* conv_w    = (const float*)d_in[2];
    const float* conv_b    = (const float*)d_in[3];
    const float* fc_w      = (const float*)d_in[4];
    const float* fc_b      = (const float*)d_in[5];
    float* out = (float*)d_out;

    char* ws = (char*)d_ws;
    float*    r  = (float*)(ws);
    float*    rw = (float*)(ws + (size_t)NB * 4096 * 4);
    _Float16* Bm = (_Float16*)(ws + (size_t)NB * 4096 * 4 + (size_t)NB * 512 * 4);

    k_route_pool<<<NB, 256, 0, stream>>>(x, conv_w, conv_b, r);
    k_route_fc<<<(NB * 512) / 256, 256, 0, stream>>>(r, fc_w, fc_b, rw);
    k_combine<<<(NB * NG * KPAD * NPAD) / 256, 256, 0, stream>>>(rw, w_experts, Bm);
    k_dyconv<<<NB * NG * 4, 128, 0, stream>>>(x, Bm, out);
}
